// MorphLayer_20409684591421
// MI455X (gfx1250) — matrix operand order repaired and checked by reading, not compiled, not run on hardware
//
#include <hip/hip_runtime.h>
#include <hip/hip_bf16.h>

typedef __attribute__((ext_vector_type(16))) __bf16 v16bf;
typedef __attribute__((ext_vector_type(8)))  float  v8f;

#define B_   32
#define C_   64
#define H_   64
#define W_   64
#define F_   128
#define HO_  62
#define WO_  62
#define WP_  66          // padded W for xT (wo up to 63 + j up to 2 -> w up to 65)
#define KTOT 576         // 9 * 64, k = (i*3+j)*64 + c
#define KSTEPS 18        // 576 / 32

// ---------------------------------------------------------------------------
// Prep 1: x (B,C,H,W) f32  ->  xT (B,H,WP,C) bf16 (channels-last, W zero-padded)
// Coalesced global read along w, LDS transpose, coalesced 32B stores along c.
// ---------------------------------------------------------------------------
__global__ __launch_bounds__(256) void xpose_kernel(const float* __restrict__ x,
                                                    __bf16* __restrict__ xT) {
  __shared__ __bf16 tile[C_][W_ + 2];   // stride 132B -> odd dword stride, conflict-friendly
  const int bh = blockIdx.x;            // b*64 + h
  const int b = bh >> 6, h = bh & 63;
  const int t = threadIdx.x;
  {
    const int c  = t >> 2;
    const int w0 = (t & 3) * 16;
    const float* src = x + (((size_t)(b * C_ + c) * H_ + h) * W_ + w0);
#pragma unroll
    for (int q = 0; q < 4; ++q) {
      float4 v = *(const float4*)(src + q * 4);
      tile[c][w0 + q * 4 + 0] = (__bf16)v.x;
      tile[c][w0 + q * 4 + 1] = (__bf16)v.y;
      tile[c][w0 + q * 4 + 2] = (__bf16)v.z;
      tile[c][w0 + q * 4 + 3] = (__bf16)v.w;
    }
  }
  __syncthreads();
  {
    const int w  = t >> 2;
    const int c0 = (t & 3) * 16;
    __align__(16) __bf16 vals[16];
#pragma unroll
    for (int cc = 0; cc < 16; ++cc) vals[cc] = tile[c0 + cc][w];
    __bf16* dst = xT + (((size_t)(b * H_ + h) * WP_ + w) * C_ + c0);
    *(uint4*)(dst)     = *(uint4*)(&vals[0]);
    *(uint4*)(dst + 8) = *(uint4*)(&vals[8]);
  }
  if (t < 128) {  // zero pad w = 64, 65
    const int w = W_ + (t >> 6);
    const int c = t & 63;
    xT[((size_t)(b * H_ + h) * WP_ + w) * C_ + c] = (__bf16)0.0f;
  }
}

// ---------------------------------------------------------------------------
// Prep 2: Wa = exp(k1) - exp(k2), bf16, pre-swizzled into a contiguous-k
// 16-bit WMMA A-operand lane layout.  Wa flat index:
//   ((ft*18 + tk)*32 + lane)*16 + p
//   row   f = ft*16 + (lane & 15)
//   wmmaK w = p + (lane>=16 ? 16 : 0)
//   k = tk*32 + w ;  k1/k2 flat = k*128 + f   (since (kh,kw,C,F) with k=(i*3+j)*64+c)
// ---------------------------------------------------------------------------
__global__ __launch_bounds__(256) void wprep_kernel(const float* __restrict__ k1,
                                                    const float* __restrict__ k2,
                                                    __bf16* __restrict__ Wa) {
  const int id = blockIdx.x * 256 + threadIdx.x;
  if (id >= KTOT * F_) return;
  const int p  = id & 15;
  const int l  = (id >> 4) & 31;
  const int g  = id >> 9;               // ft*18 + tk
  const int tk = g % KSTEPS;
  const int ft = g / KSTEPS;
  const int wk = p + ((l >> 4) << 4);
  const int k  = tk * 32 + wk;
  const int f  = ft * 16 + (l & 15);
  const int idx = k * F_ + f;
  const float v = expf(k1[idx]) - expf(k2[idx]);
  Wa[id] = (__bf16)v;
}

// ---------------------------------------------------------------------------
// Main: implicit-GEMM conv via v_wmma_f32_16x16x32_bf16, no LDS in hot loop.
// One WG per (b, ho-pair). 8 waves = 4 f-groups x 2 ho rows.
// Each wave: 32 f x 64 wo  ->  8 accumulators; per K-step: 2 A-frags + 4 B-frags,
// all via immediate-offset b128 loads off two base pointers; 8 WMMAs.
// 18 K-steps x 8 = 144 WMMAs/wave, 1.5 b128 loads per WMMA.
// ---------------------------------------------------------------------------
__global__ __launch_bounds__(256) void conv_wmma_kernel(const __bf16* __restrict__ xT,
                                                        const __bf16* __restrict__ Wa,
                                                        const float* __restrict__ bias,
                                                        float* __restrict__ out) {
  const int blk  = blockIdx.x;          // b*31 + hp
  const int b    = blk / 31;
  const int hp   = blk % 31;
  const int tid  = threadIdx.x;
  const int wave = tid >> 5;
  const int lane = tid & 31;
  const int fg    = wave & 3;           // f base = 32*fg
  const int ho    = hp * 2 + (wave >> 2);
  const int lmod  = lane & 15;          // B operand: lane column N = wo (per tile)
  const int lhalf = lane >> 4;          // 0: K 0..15, 1: K 16..31 (c-halves)

  // One base pointer each; every hot-loop load uses a compile-time offset.
  const __bf16* xbase = xT + ((size_t)(b * H_ + ho) * WP_ + lmod) * C_ + lhalf * 16;
  const __bf16* abase = Wa + ((size_t)(fg * 2) * KSTEPS * 32 + lane) * 16;

  v8f acc[8] = {v8f{}, v8f{}, v8f{}, v8f{}, v8f{}, v8f{}, v8f{}, v8f{}};

#pragma unroll
  for (int pp = 0; pp < 9; ++pp) {
    const int i = pp / 3, j = pp % 3;
#pragma unroll
    for (int ch = 0; ch < 2; ++ch) {
      const int boff = (i * WP_ + j) * C_ + ch * 32;    // elements; const after unroll
      const v16bf b0 = *(const v16bf*)(xbase + boff);
      const v16bf b1 = *(const v16bf*)(xbase + boff + 1024);   // wo tile +16
      const v16bf b2 = *(const v16bf*)(xbase + boff + 2048);   // wo tile +32
      const v16bf b3 = *(const v16bf*)(xbase + boff + 3072);   // wo tile +48
      const int tk = pp * 2 + ch;
      const v16bf a0 = *(const v16bf*)(abase + tk * 512);          // ft = fg*2
      const v16bf a1 = *(const v16bf*)(abase + tk * 512 + 9216);   // ft = fg*2+1
      acc[0] = __builtin_amdgcn_wmma_f32_16x16x32_bf16(false, a0, false, b0, (short)0, acc[0], false, false);
      acc[1] = __builtin_amdgcn_wmma_f32_16x16x32_bf16(false, a0, false, b1, (short)0, acc[1], false, false);
      acc[2] = __builtin_amdgcn_wmma_f32_16x16x32_bf16(false, a0, false, b2, (short)0, acc[2], false, false);
      acc[3] = __builtin_amdgcn_wmma_f32_16x16x32_bf16(false, a0, false, b3, (short)0, acc[3], false, false);
      acc[4] = __builtin_amdgcn_wmma_f32_16x16x32_bf16(false, a1, false, b0, (short)0, acc[4], false, false);
      acc[5] = __builtin_amdgcn_wmma_f32_16x16x32_bf16(false, a1, false, b1, (short)0, acc[5], false, false);
      acc[6] = __builtin_amdgcn_wmma_f32_16x16x32_bf16(false, a1, false, b2, (short)0, acc[6], false, false);
      acc[7] = __builtin_amdgcn_wmma_f32_16x16x32_bf16(false, a1, false, b3, (short)0, acc[7], false, false);
    }
  }

  // Epilogue: D layout -> lane = N = wo (within tile), VGPR r = M = f-offset (+8 for lanes 16..31)
  const int fb = fg * 32 + lhalf * 8;
#pragma unroll
  for (int s = 0; s < 2; ++s) {
#pragma unroll
    for (int t = 0; t < 4; ++t) {
      const int wo = t * 16 + lmod;
      if (wo < WO_) {
#pragma unroll
        for (int r = 0; r < 8; ++r) {
          const int f = fb + s * 16 + r;
          out[((size_t)(b * F_ + f) * HO_ + ho) * WO_ + wo] = acc[s * 4 + t][r] + bias[f];
        }
      }
    }
  }
}

// ---------------------------------------------------------------------------
extern "C" void kernel_launch(void* const* d_in, const int* in_sizes, int n_in,
                              void* d_out, int out_size, void* d_ws, size_t ws_size,
                              hipStream_t stream) {
  const float* x    = (const float*)d_in[0];   // (32,64,64,64)
  const float* k1   = (const float*)d_in[1];   // (3,3,64,128)
  const float* k2   = (const float*)d_in[2];   // (3,3,64,128)
  const float* bias = (const float*)d_in[3];   // (128,)
  float* out = (float*)d_out;                  // (32,128,62,62)

  char* ws = (char*)d_ws;
  const size_t xT_bytes = (size_t)B_ * H_ * WP_ * C_ * sizeof(__bf16); // 17,301,504 (256-aligned)
  __bf16* xT = (__bf16*)ws;
  __bf16* Wa = (__bf16*)(ws + xT_bytes);       // 576*128*2 = 147,456 bytes

  xpose_kernel<<<B_ * H_, 256, 0, stream>>>(x, xT);
  wprep_kernel<<<(KTOT * F_ + 255) / 256, 256, 0, stream>>>(k1, k2, Wa);
  conv_wmma_kernel<<<B_ * 31, 256, 0, stream>>>(xT, Wa, bias, out);
}
